// W8X8Linear_34883724378156
// MI455X (gfx1250) — 8-bit matrix operand order repaired and checked by reading, compiled, not run on hardware
//
#include <hip/hip_runtime.h>
#include <cstdint>

typedef __attribute__((ext_vector_type(8))) int       v8i;
typedef __attribute__((ext_vector_type(4))) int       v4i;
typedef __attribute__((ext_vector_type(8))) _Float16  v8h;

#if __has_builtin(__builtin_amdgcn_global_load_async_to_lds_b128)
#define USE_ASYNC_LDS 1
#if __has_builtin(__builtin_amdgcn_s_wait_asynccnt)
#define WAIT_ASYNC(n) __builtin_amdgcn_s_wait_asynccnt(n)
#else
#define WAIT_ASYNC(n) asm volatile("s_wait_asynccnt %0" ::"i"(n) : "memory")
#endif
#else
#define USE_ASYNC_LDS 0
#define WAIT_ASYNC(n)
#endif

#define AS1 __attribute__((address_space(1)))
#define AS3 __attribute__((address_space(3)))

__global__ __launch_bounds__(256) void quant_rows_kernel(
    const _Float16* __restrict__ src, const _Float16* __restrict__ scales,
    int8_t* __restrict__ q, _Float16* __restrict__ rmax, int K, int mode)
{
    const int lane = threadIdx.x & 31;
    const int wave = threadIdx.x >> 5;
    const long row = (long)blockIdx.x * 8 + wave;
    const _Float16* srow = src + row * (long)K;

    v8h vals[8];
    float mx = 0.0f;
#pragma unroll
    for (int i = 0; i < 8; ++i) {
        const int k = i * 256 + lane * 8;
        v8h xv = *(const v8h*)(srow + k);
        v8h sv = *(const v8h*)(scales + k);
        v8h v;
#pragma unroll
        for (int j = 0; j < 8; ++j) {
            _Float16 t = mode ? (_Float16)(xv[j] * sv[j])
                              : (_Float16)(xv[j] / sv[j]);
            v[j] = t;
            mx = fmaxf(mx, fabsf((float)t));
        }
        vals[i] = v;
    }
#pragma unroll
    for (int off = 16; off; off >>= 1)
        mx = fmaxf(mx, __shfl_xor(mx, off, 32));

    const _Float16 m16 = (_Float16)mx / (_Float16)127.0f;
    if (lane == 0) rmax[row] = m16;

#pragma unroll
    for (int i = 0; i < 8; ++i) {
        const int k = i * 256 + lane * 8;
        union { signed char c[8]; int2 p; } u;
#pragma unroll
        for (int j = 0; j < 8; ++j) {
            _Float16 qv = vals[i][j] / m16;
            u.c[j] = (signed char)(short)qv;
        }
        *(int2*)(q + row * (long)K + k) = u.p;
    }
}

#define TILE_M 128
#define TILE_N 128
#define TILE_K 64
#define LDS_STRIDE 80
#define TILE_PITCH (TILE_M * LDS_STRIDE)
#define BUF_PITCH (2 * TILE_PITCH)
#define EPI_STRIDE 144

__global__ __launch_bounds__(256) void w8a8_gemm_kernel(
    const int8_t* __restrict__ xq, const int8_t* __restrict__ wq,
    const _Float16* __restrict__ xmax, const _Float16* __restrict__ wmax,
    const _Float16* __restrict__ bias, _Float16* __restrict__ out,
    int M, int N, int K)
{
    __shared__ __align__(16) char smem[3 * BUF_PITCH];

    const int tid  = threadIdx.x;
    const int lane = tid & 31;
    const int wave = tid >> 5;
    const int wr   = wave >> 1;
    const int wc   = wave & 1;

    const int mBase = blockIdx.y * TILE_M;
    const int nBase = blockIdx.x * TILE_N;

    v8i acc[2][4] = {};

    auto copyTiles = [&](int kk, int buf) {
        char* base = smem + buf * BUF_PITCH;
#pragma unroll
        for (int i = 0; i < 2; ++i) {
            const int c   = tid + i * 256;
            const int row = c >> 2;
            const int kb  = (c & 3) * 16;
            const int8_t* ga = xq + (size_t)(mBase + row) * K + kk + kb;
            const int8_t* gb = wq + (size_t)(nBase + row) * K + kk + kb;
            char* la = base + row * LDS_STRIDE + kb;
            char* lb = base + TILE_PITCH + row * LDS_STRIDE + kb;
#if USE_ASYNC_LDS
            __builtin_amdgcn_global_load_async_to_lds_b128(
                (AS1 v4i*)ga, (AS3 v4i*)la, 0, 0);
            __builtin_amdgcn_global_load_async_to_lds_b128(
                (AS1 v4i*)gb, (AS3 v4i*)lb, 0, 0);
#else
            *(int4*)la = *(const int4*)ga;
            *(int4*)lb = *(const int4*)gb;
#endif
        }
    };

    const int KSTEPS = K / TILE_K;
    copyTiles(0, 0);
    if (KSTEPS > 1) { copyTiles(TILE_K, 1); WAIT_ASYNC(4); }
    else            { WAIT_ASYNC(0); }
    __syncthreads();

    for (int ks = 0; ks < KSTEPS; ++ks) {
        const int cur = ks % 3;
        if (ks + 2 < KSTEPS) copyTiles((ks + 2) * TILE_K, (ks + 2) % 3);

        const char* aBase = smem + cur * BUF_PITCH;
        const char* bBase = aBase + TILE_PITCH;

        v8i afrag[2], bfrag[4];
        const int aoff  = (lane < 16) ? 0 : 16;
        const int arow0 = wr * 32 + (lane & 15);
#pragma unroll
        for (int m = 0; m < 2; ++m) {
            const char* p = aBase + (arow0 + m * 16) * LDS_STRIDE + aoff;
            int4 t0 = *(const int4*)(p);
            int4 t1 = *(const int4*)(p + 32);
            v8i a = {t0.x, t0.y, t0.z, t0.w, t1.x, t1.y, t1.z, t1.w};
            afrag[m] = a;
        }
        const int boff  = (lane < 16) ? 0 : 16;
        const int bcol0 = wc * 64 + (lane & 15);
#pragma unroll
        for (int n = 0; n < 4; ++n) {
            const char* p = bBase + (bcol0 + n * 16) * LDS_STRIDE + boff;
            int4 t0 = *(const int4*)(p);
            int4 t1 = *(const int4*)(p + 32);
            v8i b = {t0.x, t0.y, t0.z, t0.w, t1.x, t1.y, t1.z, t1.w};
            bfrag[n] = b;
        }

#pragma unroll
        for (int m = 0; m < 2; ++m)
#pragma unroll
            for (int n = 0; n < 4; ++n)
                acc[m][n] = __builtin_amdgcn_wmma_i32_16x16x64_iu8(
                     true, afrag[m],  true, bfrag[n],
                    acc[m][n],  false,  false);

        if (ks + 2 < KSTEPS) WAIT_ASYNC(4);
        else                 WAIT_ASYNC(0);
        __syncthreads();
    }

    const int mWave = mBase + wr * 32;
    const int nWave = nBase + wc * 64;
    char* tile = smem + wave * (32 * EPI_STRIDE);

#pragma unroll
    for (int m = 0; m < 2; ++m) {
        const int lr0 = m * 16 + ((lane >> 4) << 3);
#pragma unroll
        for (int n = 0; n < 4; ++n) {
            const int lc       = n * 16 + (lane & 15);
            const _Float16 wmx = wmax[nWave + lc];
            const _Float16 bv  = bias[nWave + lc];
#pragma unroll
            for (int r = 0; r < 8; ++r) {
                const int lr   = lr0 + r;
                _Float16 mx    = (_Float16)(xmax[mWave + lr] * wmx);
                float    v     = (float)acc[m][n][r] * (float)mx;
                *(_Float16*)(tile + lr * EPI_STRIDE + lc * 2) =
                    (_Float16)v + bv;
            }
        }
    }
    const size_t gRowOff = ((size_t)(mWave + lane) * N + nWave) * 2;
    const char*  src     = tile + lane * EPI_STRIDE;
#pragma unroll
    for (int i = 0; i < 8; ++i) {
        int4 v = *(const int4*)(src + i * 16);
        *(int4*)((char*)out + gRowOff + i * 16) = v;
    }
}

extern "C" void kernel_launch(void* const* d_in, const int* in_sizes, int n_in,
                              void* d_out, int out_size, void* d_ws, size_t ws_size,
                              hipStream_t stream)
{
    const _Float16* x      = (const _Float16*)d_in[0];
    const _Float16* ori_w  = (const _Float16*)d_in[1];
    const _Float16* scales = (const _Float16*)d_in[2];
    const _Float16* bias   = (const _Float16*)d_in[3];
    _Float16*       out    = (_Float16*)d_out;

    const int K = in_sizes[2];
    const int N = in_sizes[3];
    const int M = in_sizes[0] / K;

    char* ws = (char*)d_ws;
    int8_t*   xq   = (int8_t*)ws;
    int8_t*   wq   = (int8_t*)(ws + (size_t)M * K);
    _Float16* xmax = (_Float16*)(ws + (size_t)M * K + (size_t)N * K);
    _Float16* wmax = xmax + M;

    quant_rows_kernel<<<N / 8, 256, 0, stream>>>(ori_w, scales, wq, wmax, K, 1);
    quant_rows_kernel<<<M / 8, 256, 0, stream>>>(x, scales, xq, xmax, K, 0);

    dim3 grid(N / TILE_N, M / TILE_M);
    w8a8_gemm_kernel<<<grid, 256, 0, stream>>>(xq, wq, xmax, wmax, bias, out,
                                               M, N, K);
}
